// MultiHeadAttention_56143812493747
// MI455X (gfx1250) — matrix operand order repaired and checked by reading, not compiled, not run on hardware
//
#include <hip/hip_runtime.h>
#include <hip/hip_bf16.h>
#include <math.h>

// ---------------------------------------------------------------------------
// MI455X (gfx1250) multi-head attention block.
// wave32 + v_wmma_f32_16x16x32_f16 + async global->LDS double buffering.
// Shapes: B=16, C=512, H=W=32 (T=1024), groups=32, heads=8, hd=64.
//
// WMMA B-operands (xn, k, v, h) are stored pre-swizzled in B-fragment order:
//   buf[((kt*nTiles + nt)*32 + lane)*16 + j]
//   k = kt*32 + (lane>>4)*16 + j,  n = nt*16 + (lane&15)
// so a per-K-step B slab is one contiguous block -> bulk async copy to LDS.
// ---------------------------------------------------------------------------

#define B_   16
#define C_   512
#define T_   1024
#define NH_  8
#define HD_  64
#define BH_  (B_ * NH_)
#define SCALE_QK 0.3535533905932738f   // 64^-0.25

typedef _Float16 v16h __attribute__((ext_vector_type(16)));
typedef _Float16 v8h  __attribute__((ext_vector_type(8)));
typedef float    v8f  __attribute__((ext_vector_type(8)));

#define LDS_AS __attribute__((address_space(3)))
#define SCHED_FENCE() __builtin_amdgcn_sched_barrier(0)

__device__ __forceinline__ unsigned lds_byte_off(const void* p) {
  return (unsigned)(size_t)(LDS_AS const char*)p;
}

// async global -> LDS copy, 16B per lane, tracked by ASYNCcnt
__device__ __forceinline__ void async_b128(unsigned lds_off, const void* g) {
  asm volatile("global_load_async_to_lds_b128 %0, %1, off"
               :: "v"(lds_off), "v"((unsigned long long)(size_t)g)
               : "memory");
}

template <int N>
__device__ __forceinline__ void wait_async() {
  asm volatile("s_wait_asynccnt %0" :: "i"(N) : "memory");
}

__device__ __forceinline__ v8f zero8() {
  v8f z;
#pragma unroll
  for (int i = 0; i < 8; ++i) z[i] = 0.0f;
  return z;
}

// A-fragment: 16x32 f16, M = lane%16; two 16B vector loads per lane.
__device__ __forceinline__ v16h load_a_frag(const _Float16* src, int lda) {
  const int lane = threadIdx.x & 31;
  const int row  = lane & 15;
  const int koff = (lane >> 4) << 4;           // 0 or 16
  const _Float16* p = src + row * lda + koff;
  v8h lo = *reinterpret_cast<const v8h*>(p);        // K = koff + 0..7
  v8h hi = *reinterpret_cast<const v8h*>(p + 8);    // K = koff + 8..15
  v16h a;
#pragma unroll
  for (int j = 0; j < 8; ++j) { a[j] = lo[j]; a[j + 8] = hi[j]; }
  return a;
}

// B-fragment from a staged LDS slab: 32B contiguous per lane.
__device__ __forceinline__ v16h load_b_lds(const _Float16* base, int chunk) {
  const int lane = threadIdx.x & 31;
  return *reinterpret_cast<const v16h*>(base + ((chunk * 32 + lane) << 4));
}

// scalar index into swizzled B storage for element (k, n)
__device__ __forceinline__ size_t bswz(int k, int n, int nTiles) {
  return ((((size_t)(k >> 5) * nTiles + (n >> 4)) * 32 +
           ((((k >> 4) & 1) << 4) | (n & 15))) << 4) + (k & 15);
}

__device__ __forceinline__ v8f wmma16(v16h a, v16h b, v8f c) {
  return __builtin_amdgcn_wmma_f32_16x16x32_f16(false, a, false, b,
                                                (short)0, c, false, false);
}

// ---------------------------------------------------------------------------
// Kernel 0: f32 -> f16 conversion (weights, row-major A operands)
// ---------------------------------------------------------------------------
__global__ void cvt_f32_to_f16(const float* __restrict__ src,
                               _Float16* __restrict__ dst, int n) {
  int i = blockIdx.x * blockDim.x + threadIdx.x;
  if (i < n) dst[i] = (_Float16)src[i];
}

// ---------------------------------------------------------------------------
// Kernel 1: GroupNorm -> f16, stored B-swizzled over (K=C, N=T).
// ---------------------------------------------------------------------------
__global__ void groupnorm_kernel(const float* __restrict__ x,
                                 const float* __restrict__ w,
                                 const float* __restrict__ bias,
                                 _Float16* __restrict__ xn_swz) {
  const int b = blockIdx.x >> 5;
  const int g = blockIdx.x & 31;
  const int tid = threadIdx.x;
  const int N = 16 * T_;
  const float* base = x + ((size_t)b * C_ + g * 16) * T_;

  float s = 0.0f, s2 = 0.0f;
  for (int i = tid; i < N; i += 256) {
    float v = base[i];
    s += v; s2 += v * v;
  }
  __shared__ float red0[256], red1[256];
  red0[tid] = s; red1[tid] = s2;
  __syncthreads();
  for (int st = 128; st > 0; st >>= 1) {
    if (tid < st) { red0[tid] += red0[tid + st]; red1[tid] += red1[tid + st]; }
    __syncthreads();
  }
  const float mean = red0[0] * (1.0f / N);
  const float var  = red1[0] * (1.0f / N) - mean * mean;
  const float rstd = rsqrtf(var + 1e-5f);

  float wl[16], bl[16];
#pragma unroll
  for (int cl = 0; cl < 16; ++cl) {
    wl[cl] = w[g * 16 + cl] * rstd;
    bl[cl] = bias[g * 16 + cl];
  }

  _Float16* xnb = xn_swz + (size_t)b * C_ * T_;
  const int kt = g >> 1;
  const int khalf = (g & 1) << 4;
  for (int t = tid; t < T_; t += 256) {
    v16h pk;
#pragma unroll
    for (int cl = 0; cl < 16; ++cl) {
      float v = (base[(size_t)cl * T_ + t] - mean) * wl[cl] + bl[cl];
      pk[cl] = (_Float16)v;
    }
    size_t idx = (((size_t)kt * 64 + (t >> 4)) * 32 + (khalf | (t & 15))) << 4;
    *reinterpret_cast<v16h*>(xnb + idx) = pk;
  }
}

// ---------------------------------------------------------------------------
// Kernel 2: QKV GEMM with async-LDS double-buffered B + pipelined A.
// Block = 8 waves = 128(M) x 128(N); per K-step B slab = contiguous 8KB.
// ---------------------------------------------------------------------------
__global__ void qkv_gemm_kernel(const _Float16* __restrict__ Wq,
                                const _Float16* __restrict__ xn_swz,
                                const float* __restrict__ qkv_bias,
                                _Float16* __restrict__ q_t,
                                _Float16* __restrict__ k_swz,
                                _Float16* __restrict__ v_swz) {
  __shared__ __align__(32) _Float16 ldsB[2][128 * 32];   // 8 KB per buffer
  const int tid  = threadIdx.x;
  const int wv   = tid >> 5;
  const int lane = tid & 31;
  const int oBase = blockIdx.x * 128 + wv * 16;   // 0..1535
  const int tBase = blockIdx.y * 128;
  const int nt0   = tBase >> 4;
  const int b     = blockIdx.z;

  const _Float16* A  = Wq + (size_t)oBase * C_;
  const _Float16* Bb = xn_swz + (size_t)b * C_ * T_;

  auto stage = [&](int kt, int buf) {
    const char* src =
        (const char*)(Bb + ((((size_t)kt * (T_ / 16) + nt0) * 32) << 4));
    unsigned dst = lds_byte_off(&ldsB[buf][0]);
    async_b128(dst + tid * 16,        src + tid * 16);
    async_b128(dst + 4096 + tid * 16, src + 4096 + tid * 16);
  };

  v8f acc[8];
#pragma unroll
  for (int i = 0; i < 8; ++i) acc[i] = zero8();

  stage(0, 0);
  v16h af = load_a_frag(A, C_);               // A tile kb=0
  int cur = 0;
  for (int kb = 0; kb < C_; kb += 32) {
    const bool has_next = (kb + 32 < C_);
    if (has_next) {
      stage((kb >> 5) + 1, cur ^ 1);
      wait_async<2>();           // current tile landed, next in flight
    } else {
      wait_async<0>();
    }
    __syncthreads();
    const _Float16* bp = &ldsB[cur][0];
    // next A-tile global load issued and PINNED above the WMMA burst
    v16h afn;
    if (has_next) afn = load_a_frag(A + kb + 32, C_);
    v16h bcur = load_b_lds(bp, 0);
    SCHED_FENCE();
    // depth-1 rotating pipeline: load frag nt+1, fence, WMMA frag nt
#pragma unroll
    for (int nt = 0; nt < 8; ++nt) {
      v16h bnext;
      if (nt < 7) bnext = load_b_lds(bp, nt + 1);
      SCHED_FENCE();
      acc[nt] = wmma16(af, bcur, acc[nt]);
      bcur = bnext;
    }
    __syncthreads();
    af = afn;
    cur ^= 1;
  }

  const int lh = lane >> 4, ln = lane & 15;
  float biasr[8];
#pragma unroll
  for (int r = 0; r < 8; ++r) biasr[r] = qkv_bias[oBase + r + (lh << 3)];

  if (oBase < C_) {                               // ---- Q (t-major rows)
    const int head = oBase >> 6, c0 = oBase & 63;
    _Float16* qh = q_t + (size_t)(b * NH_ + head) * T_ * HD_;
#pragma unroll
    for (int nt = 0; nt < 8; ++nt) {
      const int t = tBase + nt * 16 + ln;
      v8h pk;
#pragma unroll
      for (int r = 0; r < 8; ++r)
        pk[r] = (_Float16)((acc[nt][r] + biasr[r]) * SCALE_QK);
      *reinterpret_cast<v8h*>(qh + (size_t)t * HD_ + c0 + (lh << 3)) = pk;
    }
  } else if (oBase < 2 * C_) {                    // ---- K (B-swizzled)
    const int c0 = oBase - C_;
    const int head = c0 >> 6, cl0 = c0 & 63;
    _Float16* kh = k_swz + (size_t)(b * NH_ + head) * HD_ * T_;
    const int kt = cl0 >> 5;
    const int swzlane = (((cl0 >> 4) & 1) << 4) | ln;
#pragma unroll
    for (int nt = 0; nt < 8; ++nt) {
      v8h pk;
#pragma unroll
      for (int r = 0; r < 8; ++r)
        pk[r] = (_Float16)((acc[nt][r] + biasr[r]) * SCALE_QK);
      const int ntile = nt0 + nt;
      *reinterpret_cast<v8h*>(
          kh + ((((size_t)kt * (T_ / 16) + ntile) * 32 + swzlane) << 4) +
          (lh << 3)) = pk;
    }
  } else {                                        // ---- V (B-swizzled, K=s)
    const int c0 = oBase - 2 * C_;
    const int head = c0 >> 6, cl0 = c0 & 63;
    _Float16* vh = v_swz + (size_t)(b * NH_ + head) * T_ * HD_;
#pragma unroll
    for (int nt = 0; nt < 8; ++nt) {
      const int t = tBase + nt * 16 + ln;         // K index (s)
#pragma unroll
      for (int r = 0; r < 8; ++r) {
        const int c = cl0 + r + (lh << 3);        // N index (c)
        vh[bswz(t, c, HD_ / 16)] = (_Float16)(acc[nt][r] + biasr[r]);
      }
    }
  }
}

// ---------------------------------------------------------------------------
// Kernel 3: flash attention, block-synchronous, async-LDS K/V double buffer.
// Block = 8 waves; each wave owns a 16-row t tile of the same (bh).
// ---------------------------------------------------------------------------
__global__ void flash_attn_kernel(const _Float16* __restrict__ q_t,
                                  const _Float16* __restrict__ k_swz,
                                  const _Float16* __restrict__ v_swz,
                                  _Float16* __restrict__ h_swz) {
  __shared__ __align__(32) _Float16 ldsK[2][2048];      // 4 KB each
  __shared__ __align__(32) _Float16 ldsV[2][2048];      // 4 KB each
  __shared__ __align__(16) _Float16 ldsP[8][16 * 32];   // per-wave P tile
  const int tid  = threadIdx.x;
  const int wv   = tid >> 5;
  const int lane = tid & 31;
  const int bh   = blockIdx.y;
  const int t0   = blockIdx.x * 128 + wv * 16;
  const int bB   = bh >> 3, head = bh & 7;

  const _Float16* q  = q_t + ((size_t)bh * T_ + t0) * HD_;
  const _Float16* kk = k_swz + (size_t)bh * HD_ * T_;
  const _Float16* vv = v_swz + (size_t)bh * T_ * HD_;

  // K s-block slab: kt in {0,1} x 2 n-tiles -> two contiguous 2KB regions.
  auto stageK = [&](int st, int buf) {
    const int kt = tid >> 7, i = tid & 127;
    const char* src =
        (const char*)(kk + ((((size_t)kt * (T_ / 16) + st) * 32) << 4)) + i * 16;
    async_b128(lds_byte_off(&ldsK[buf][0]) + kt * 2048 + i * 16, src);
  };
  // V s-block slab: one contiguous 4KB region.
  auto stageV = [&](int vkt, int buf) {
    const char* src = (const char*)(vv + (((size_t)vkt * 4 * 32) << 4)) + tid * 16;
    async_b128(lds_byte_off(&ldsV[buf][0]) + tid * 16, src);
  };

  const v16h aq0 = load_a_frag(q, HD_);        // K = c 0..31
  const v16h aq1 = load_a_frag(q + 32, HD_);   // K = c 32..63

  v8f acc[4];
#pragma unroll
  for (int i = 0; i < 4; ++i) acc[i] = zero8();
  float m_run[8], l_run[8];
#pragma unroll
  for (int r = 0; r < 8; ++r) { m_run[r] = -INFINITY; l_run[r] = 0.0f; }

  const int lh = lane >> 4;
  _Float16* P = &ldsP[wv][0];

  stageK(0, 0);
  stageV(0, 0);
  int cur = 0;
  for (int s0 = 0; s0 < T_; s0 += 32) {
    if (s0 + 32 < T_) {
      stageK((s0 >> 4) + 2, cur ^ 1);
      stageV((s0 >> 5) + 1, cur ^ 1);
      wait_async<2>();           // current s-block landed, next in flight
    } else {
      wait_async<0>();
    }
    __syncthreads();

    // all 4 K fragments issued together -> staggered dscnt waits
    const _Float16* kp = &ldsK[cur][0];
    v16h bk0 = load_b_lds(kp, 0);
    v16h bk1 = load_b_lds(kp, 1);
    v16h bk2 = load_b_lds(kp, 2);
    v16h bk3 = load_b_lds(kp, 3);
    SCHED_FENCE();
    v8f sc0 = zero8(), sc1 = zero8();
    sc0 = wmma16(aq0, bk0, sc0);
    sc1 = wmma16(aq0, bk1, sc1);
    sc0 = wmma16(aq1, bk2, sc0);
    sc1 = wmma16(aq1, bk3, sc1);

    // V fragments pinned here: latency hidden under the softmax VALU burst
    const _Float16* vp = &ldsV[cur][0];
    v16h bv0 = load_b_lds(vp, 0);
    v16h bv1 = load_b_lds(vp, 1);
    v16h bv2 = load_b_lds(vp, 2);
    v16h bv3 = load_b_lds(vp, 3);
    SCHED_FENCE();

    // online softmax per row (row lives in a 16-lane group)
#pragma unroll
    for (int r = 0; r < 8; ++r) {
      float mx = fmaxf(sc0[r], sc1[r]);
      mx = fmaxf(mx, __shfl_xor(mx, 1, 16));
      mx = fmaxf(mx, __shfl_xor(mx, 2, 16));
      mx = fmaxf(mx, __shfl_xor(mx, 4, 16));
      mx = fmaxf(mx, __shfl_xor(mx, 8, 16));
      float mnew  = fmaxf(m_run[r], mx);
      float alpha = __expf(m_run[r] - mnew);
      float p0 = __expf(sc0[r] - mnew);
      float p1 = __expf(sc1[r] - mnew);
      float rs = p0 + p1;
      rs += __shfl_xor(rs, 1, 16);
      rs += __shfl_xor(rs, 2, 16);
      rs += __shfl_xor(rs, 4, 16);
      rs += __shfl_xor(rs, 8, 16);
      l_run[r] = l_run[r] * alpha + rs;
      m_run[r] = mnew;
#pragma unroll
      for (int nt = 0; nt < 4; ++nt) acc[nt][r] *= alpha;
      const int m = r + (lh << 3);
      P[m * 32 + (lane & 15)]      = (_Float16)p0;
      P[m * 32 + 16 + (lane & 15)] = (_Float16)p1;
    }

    // P (16x32) @ V (32x64); P is wave-private LDS (DS in-order per wave)
    v16h ap = load_a_frag(P, 32);
    acc[0] = wmma16(ap, bv0, acc[0]);
    acc[1] = wmma16(ap, bv1, acc[1]);
    acc[2] = wmma16(ap, bv2, acc[2]);
    acc[3] = wmma16(ap, bv3, acc[3]);

    __syncthreads();             // all waves done reading before overwrite
    cur ^= 1;
  }

  // normalize; store B-swizzled over (K=c 512, N=t 1024) for the proj GEMM
  _Float16* hb = h_swz + (size_t)bB * C_ * T_;
#pragma unroll
  for (int r = 0; r < 8; ++r) {
    const float inv = 1.0f / l_run[r];
    const int t = t0 + r + (lh << 3);
#pragma unroll
    for (int nt = 0; nt < 4; ++nt) {
      const int c = head * HD_ + nt * 16 + (lane & 15);
      hb[bswz(c, t, T_ / 16)] = (_Float16)(acc[nt][r] * inv);
    }
  }
}

// ---------------------------------------------------------------------------
// Kernel 4: proj GEMM (async-LDS B, pipelined A) + bias + residual -> f32.
// ---------------------------------------------------------------------------
__global__ void proj_gemm_kernel(const _Float16* __restrict__ Wp,
                                 const _Float16* __restrict__ h_swz,
                                 const float* __restrict__ proj_bias,
                                 const float* __restrict__ x,
                                 float* __restrict__ out) {
  __shared__ __align__(32) _Float16 ldsB[2][128 * 32];
  const int tid  = threadIdx.x;
  const int wv   = tid >> 5;
  const int lane = tid & 31;
  const int oBase = blockIdx.x * 128 + wv * 16;   // 0..511
  const int tBase = blockIdx.y * 128;
  const int nt0   = tBase >> 4;
  const int b     = blockIdx.z;

  const _Float16* A  = Wp + (size_t)oBase * C_;
  const _Float16* Bb = h_swz + (size_t)b * C_ * T_;

  auto stage = [&](int kt, int buf) {
    const char* src =
        (const char*)(Bb + ((((size_t)kt * (T_ / 16) + nt0) * 32) << 4));
    unsigned dst = lds_byte_off(&ldsB[buf][0]);
    async_b128(dst + tid * 16,        src + tid * 16);
    async_b128(dst + 4096 + tid * 16, src + 4096 + tid * 16);
  };

  v8f acc[8];
#pragma unroll
  for (int i = 0; i < 8; ++i) acc[i] = zero8();

  stage(0, 0);
  v16h af = load_a_frag(A, C_);
  int cur = 0;
  for (int kb = 0; kb < C_; kb += 32) {
    const bool has_next = (kb + 32 < C_);
    if (has_next) {
      stage((kb >> 5) + 1, cur ^ 1);
      wait_async<2>();
    } else {
      wait_async<0>();
    }
    __syncthreads();
    const _Float16* bp = &ldsB[cur][0];
    v16h afn;
    if (has_next) afn = load_a_frag(A + kb + 32, C_);
    v16h bcur = load_b_lds(bp, 0);
    SCHED_FENCE();
#pragma unroll
    for (int nt = 0; nt < 8; ++nt) {
      v16h bnext;
      if (nt < 7) bnext = load_b_lds(bp, nt + 1);
      SCHED_FENCE();
      acc[nt] = wmma16(af, bcur, acc[nt]);
      bcur = bnext;
    }
    __syncthreads();
    af = afn;
    cur ^= 1;
  }

  const int lh = lane >> 4, ln = lane & 15;
#pragma unroll
  for (int r = 0; r < 8; ++r) {
    const int o = oBase + r + (lh << 3);
    const float pb = proj_bias[o];
#pragma unroll
    for (int nt = 0; nt < 8; ++nt) {
      const int t = tBase + nt * 16 + ln;
      const size_t idx = ((size_t)b * C_ + o) * T_ + t;
      out[idx] = acc[nt][r] + pb + x[idx];
    }
  }
}

// ---------------------------------------------------------------------------
// Launch
// ---------------------------------------------------------------------------
extern "C" void kernel_launch(void* const* d_in, const int* in_sizes, int n_in,
                              void* d_out, int out_size, void* d_ws, size_t ws_size,
                              hipStream_t stream) {
  const float* x      = (const float*)d_in[0];
  const float* norm_w = (const float*)d_in[1];
  const float* norm_b = (const float*)d_in[2];
  const float* qkv_w  = (const float*)d_in[3];
  const float* qkv_b  = (const float*)d_in[4];
  const float* proj_w = (const float*)d_in[5];
  const float* proj_b = (const float*)d_in[6];
  // d_in[7] = num_heads (== 8, hardcoded)
  float* out = (float*)d_out;

  char* ws = (char*)d_ws;
  size_t off = 0;
  auto alloc = [&](size_t bytes) -> char* {
    char* p = ws + off;
    off += (bytes + 255) & ~(size_t)255;
    return p;
  };
  _Float16* wq16  = (_Float16*)alloc((size_t)3 * C_ * C_ * 2);
  _Float16* wp16  = (_Float16*)alloc((size_t)C_ * C_ * 2);
  _Float16* xn    = (_Float16*)alloc((size_t)B_ * C_ * T_ * 2);   // B-swizzled
  _Float16* q_t   = (_Float16*)alloc((size_t)BH_ * T_ * HD_ * 2); // row-major
  _Float16* k_swz = (_Float16*)alloc((size_t)BH_ * HD_ * T_ * 2); // B-swizzled
  _Float16* v_swz = (_Float16*)alloc((size_t)BH_ * T_ * HD_ * 2); // B-swizzled
  _Float16* h_swz = (_Float16*)alloc((size_t)B_ * C_ * T_ * 2);   // B-swizzled
  (void)ws_size; (void)in_sizes; (void)n_in; (void)out_size;

  const int nWq = 3 * C_ * C_, nWp = C_ * C_;
  cvt_f32_to_f16<<<(nWq + 255) / 256, 256, 0, stream>>>(qkv_w, wq16, nWq);
  cvt_f32_to_f16<<<(nWp + 255) / 256, 256, 0, stream>>>(proj_w, wp16, nWp);

  groupnorm_kernel<<<B_ * 32, 256, 0, stream>>>(x, norm_w, norm_b, xn);

  qkv_gemm_kernel<<<dim3(3 * C_ / 128, T_ / 128, B_), 256, 0, stream>>>(
      wq16, xn, qkv_b, q_t, k_swz, v_swz);

  flash_attn_kernel<<<dim3(T_ / 128, BH_), 256, 0, stream>>>(q_t, k_swz, v_swz,
                                                             h_swz);

  proj_gemm_kernel<<<dim3(C_ / 128, T_ / 128, B_), 256, 0, stream>>>(
      wp16, h_swz, proj_b, x, out);
}
